// Encoder_12704513261661
// MI455X (gfx1250) — matrix operand order repaired and checked by reading, not compiled, not run on hardware
//
#include <hip/hip_runtime.h>
#include <hip/hip_bf16.h>
#include <math.h>

typedef __attribute__((ext_vector_type(16))) __bf16 v16bf;
typedef __attribute__((ext_vector_type(8)))  __bf16 v8bf;
typedef __attribute__((ext_vector_type(8)))  float   v8f;
typedef __attribute__((ext_vector_type(4))) unsigned int u32x4;
typedef __attribute__((ext_vector_type(8))) int         i32x8;
typedef __attribute__((ext_vector_type(4))) int         i32x4;

constexpr int kB  = 2;
constexpr int kD  = 1024;
constexpr int kH  = 16;
constexpr int kDH = 64;
constexpr int kFF = 4096;
constexpr float kEps = 1e-5f;

#if __has_builtin(__builtin_amdgcn_tensor_load_to_lds) && __has_builtin(__builtin_amdgcn_s_wait_tensorcnt)
#define USE_TDM 1
#else
#define USE_TDM 0
#endif

// ---- DPP row_xmask lane exchange within 16-lane rows (no LDS, no DScnt wait)
template <int XMASK>
__device__ __forceinline__ float dpp_xor(float v) {
#if __has_builtin(__builtin_amdgcn_update_dpp)
  const int r = __builtin_amdgcn_update_dpp(0, __float_as_int(v),
                                            0x160 | XMASK, 0xf, 0xf, true);
  return __int_as_float(r);
#else
  return __shfl_xor(v, XMASK, 32);
#endif
}
__device__ __forceinline__ float row16_max(float v) {
  v = fmaxf(v, dpp_xor<1>(v));
  v = fmaxf(v, dpp_xor<2>(v));
  v = fmaxf(v, dpp_xor<4>(v));
  v = fmaxf(v, dpp_xor<8>(v));
  return v;
}
__device__ __forceinline__ float row16_sum(float v) {
  v += dpp_xor<1>(v);
  v += dpp_xor<2>(v);
  v += dpp_xor<4>(v);
  v += dpp_xor<8>(v);
  return v;
}

#if USE_TDM
// Build a 2D-tile TDM descriptor (D#) and issue TENSOR_LOAD_TO_LDS.
// Group0: count=1 | lds_addr | global_addr(57b) | type=2
// Group1: data_size=2B | tensor_dim0/1 | tile_dim0/1 | tensor_dim0_stride(48b)
__device__ __forceinline__ void tdm_load_2d(unsigned lds_byte_off, const void* gsrc,
                                            unsigned dim0, unsigned dim1,
                                            unsigned tile0, unsigned tile1,
                                            unsigned long long stride0) {
  const unsigned long long ga = (unsigned long long)(uintptr_t)gsrc;
  u32x4 g0;
  g0[0] = 1u;                                            // count=1, user desc
  g0[1] = lds_byte_off;                                  // lds_addr
  g0[2] = (unsigned)(ga & 0xFFFFFFFFu);                  // global_addr[31:0]
  g0[3] = (unsigned)((ga >> 32) & 0x01FFFFFFu) | (2u << 30); // addr[56:32] | type=2
  i32x8 g1;
  g1[0] = (int)(1u << 16);                               // data_size=1 -> 2 bytes
  g1[1] = (int)((dim0 & 0xFFFFu) << 16);                 // tensor_dim0 lo16 (bits 63:48)
  g1[2] = (int)(((dim0 >> 16) & 0xFFFFu) | ((dim1 & 0xFFFFu) << 16));
  g1[3] = (int)(((dim1 >> 16) & 0xFFFFu) | ((tile0 & 0xFFFFu) << 16));
  g1[4] = (int)(tile1 & 0xFFFFu);                        // tile_dim1, tile_dim2=0
  g1[5] = (int)(stride0 & 0xFFFFFFFFull);                // dim0_stride lo32
  g1[6] = (int)((stride0 >> 32) & 0xFFFFull);            // dim0_stride hi16
  g1[7] = 0;
  const i32x4 z4 = {0, 0, 0, 0};
#if defined(__clang_major__) && (__clang_major__ >= 23)
  const i32x8 z8 = {0, 0, 0, 0, 0, 0, 0, 0};
  __builtin_amdgcn_tensor_load_to_lds(g0, g1, z4, z4, z8, 0);
#else
  __builtin_amdgcn_tensor_load_to_lds(g0, g1, z4, z4, 0);
#endif
}
#endif

__device__ __forceinline__ unsigned lds_off_of(const void* p) {
  // LDS aperture: low 32 bits of the flat address are the LDS byte offset.
  return (unsigned)(uintptr_t)p;
}

// ---------------------------------------------------------------- conversion
__global__ void f32_to_bf16_kernel(const float* __restrict__ in,
                                   __bf16* __restrict__ out, int n) {
  int i = blockIdx.x * blockDim.x + threadIdx.x;
  int stride = gridDim.x * blockDim.x;
  for (; i < n; i += stride) out[i] = (__bf16)in[i];
}

// ------------------------------------------------------- embedding + posenc
__global__ void embed_pe_kernel(const int* __restrict__ tokens,
                                const float* __restrict__ emb,
                                float* __restrict__ xf, __bf16* __restrict__ xb,
                                int S) {
  const int row = blockIdx.x;           // b*S + s
  const int s   = row % S;
  const int tok = tokens[row];
  const float* erow = emb + (size_t)tok * kD;
  const float inv = -logf(10000.0f) / (float)kD;
  for (int d = threadIdx.x; d < kD; d += blockDim.x) {
    float div = __expf((float)(d & ~1) * inv);
    float ang = (float)s * div;
    float pe  = (d & 1) ? cosf(ang) : sinf(ang);
    float v   = erow[d] + pe;
    xf[(size_t)row * kD + d] = v;
    xb[(size_t)row * kD + d] = (__bf16)v;
  }
}

// ---------------------------------------------------------- WMMA GEMM (TDM)
// C[M,N] = op(A[M,K] @ B[K,N] + bias).  Block = 256 threads = 8 waves.
// Block tile: 64(M) x 128(N); wave -> 16(M) x 64(N) (4 accumulators).
// A/B 32-K tiles staged into double-buffered LDS by the Tensor Data Mover.
template <typename OutT, bool BIAS, bool RELU>
__global__ void __launch_bounds__(256)
gemm_wmma_bf16(const __bf16* __restrict__ A, const __bf16* __restrict__ B,
               OutT* __restrict__ C, const float* __restrict__ bias,
               int M, int N, int K) {
  __shared__ __bf16 ldsA[2][64][32];     // [buf][m][k]
  __shared__ __bf16 ldsB[2][32][128];    // [buf][k][n]

  const int lane = threadIdx.x & 31;
  const int wv   = threadIdx.x >> 5;
  const int half = lane >> 4;
  const int l16  = lane & 15;
  const int wm   = wv & 3;               // m-tile within block (0..3)
  const int wn   = wv >> 2;              // n-half within block (0..1)
  const int blk_m = blockIdx.y * 64;
  const int blk_n = blockIdx.x * 128;

  const int kbA = half * 16;             // A-frag k-base
  const int kbB = half * 16;             // B-frag k-base

  auto stage = [&](int buf, int k0) {
#if USE_TDM
    if (wv == 0) {
      tdm_load_2d(lds_off_of(&ldsA[buf][0][0]),
                  A + (size_t)blk_m * K + k0,
                  (unsigned)(K - k0), (unsigned)(M - blk_m),
                  32u, 64u, (unsigned long long)K);
      tdm_load_2d(lds_off_of(&ldsB[buf][0][0]),
                  B + (size_t)k0 * N + blk_n,
                  (unsigned)(N - blk_n), (unsigned)(K - k0),
                  128u, 32u, (unsigned long long)N);
    }
#else
    for (int c = threadIdx.x; c < 64 * 4; c += 256) {       // A: 64 rows x 4 chunks
      const int row = c >> 2, col8 = (c & 3) * 8;
      *(v8bf*)&ldsA[buf][row][col8] =
          *(const v8bf*)(A + (size_t)(blk_m + row) * K + k0 + col8);
    }
    for (int c = threadIdx.x; c < 32 * 16; c += 256) {      // B: 32 rows x 16 chunks
      const int row = c >> 4, col8 = (c & 15) * 8;
      *(v8bf*)&ldsB[buf][row][col8] =
          *(const v8bf*)(B + (size_t)(k0 + row) * N + blk_n + col8);
    }
#endif
  };

  v8f acc[4];
#pragma unroll
  for (int t = 0; t < 4; ++t)
#pragma unroll
    for (int r = 0; r < 8; ++r) acc[t][r] = 0.0f;

  stage(0, 0);
  for (int k0 = 0; k0 < K; k0 += 32) {
    const int buf = (k0 >> 5) & 1;
    if (k0 + 32 < K) {
      stage(buf ^ 1, k0 + 32);
#if USE_TDM
      __builtin_amdgcn_s_wait_tensorcnt(1);   // oldest (current buf) complete
#endif
    } else {
#if USE_TDM
      __builtin_amdgcn_s_wait_tensorcnt(0);
#endif
    }
    __syncthreads();

    // A fragment: row (wm*16 + l16), contiguous 8+8 elems from LDS
    v16bf a;
    {
      const __bf16* arow = &ldsA[buf][wm * 16 + l16][0];
      const v8bf lo = *(const v8bf*)(arow + kbA);
      const v8bf hi = *(const v8bf*)(arow + kbA + 8);
#pragma unroll
      for (int e = 0; e < 8; ++e) { a[e] = lo[e]; a[8 + e] = hi[e]; }
    }
    // 4 B fragments / WMMAs sharing the A fragment
#pragma unroll
    for (int t = 0; t < 4; ++t) {
      const int ncol = wn * 64 + t * 16 + l16;
      v16bf bfrag;
#pragma unroll
      for (int e = 0; e < 16; ++e) bfrag[e] = ldsB[buf][kbB + e][ncol];
      acc[t] = __builtin_amdgcn_wmma_f32_16x16x32_bf16(
          false, a, false, bfrag, (short)0, acc[t], false, false);
    }
    __syncthreads();
  }

  // store
#pragma unroll
  for (int t = 0; t < 4; ++t) {
    const int n = blk_n + wn * 64 + t * 16 + l16;
    const float bvv = BIAS ? bias[n] : 0.0f;
#pragma unroll
    for (int r = 0; r < 8; ++r) {
      const int m = blk_m + wm * 16 + r + half * 8;
      float v = acc[t][r] + bvv;
      if (RELU) v = fmaxf(v, 0.0f);
      C[(size_t)m * N + n] = (OutT)v;
    }
  }
}

// ------------------------------------------------- fused flash-style attention
// One wave per (b, h, 16-query tile); streams keys 32 at a time.
__global__ void __launch_bounds__(256)
attn_kernel(const __bf16* __restrict__ Q, const __bf16* __restrict__ K,
            const __bf16* __restrict__ V, __bf16* __restrict__ O, int S) {
  __shared__ __bf16 ldsP[8][16][32];   // per-wave P tile (16 q x 32 k)
  const int lane = threadIdx.x & 31;
  const int wv   = threadIdx.x >> 5;
  const int half = lane >> 4;
  const int l16  = lane & 15;

  int wid = blockIdx.x * 8 + wv;
  const int ntq = S >> 4;
  const int qt = wid % ntq; wid /= ntq;
  const int h  = wid % kH;  wid /= kH;
  const int b  = wid;

  const int qrow = qt * 16 + l16;
  const __bf16* qptr = Q + ((size_t)(b * S + qrow) * kD) + h * kDH;

  // Q A-fragments for the two 32-wide K-dim chunks of DH=64
  v16bf aq[2];
#pragma unroll
  for (int c = 0; c < 2; ++c) {
    const v8bf lo = *(const v8bf*)(qptr + c * 32 + half * 16);
    const v8bf hi = *(const v8bf*)(qptr + c * 32 + half * 16 + 8);
#pragma unroll
    for (int e = 0; e < 8; ++e) { aq[c][e] = lo[e]; aq[c][8 + e] = hi[e]; }
  }

  v8f o_acc[4];
  float rm[8], rs[8];
#pragma unroll
  for (int t = 0; t < 4; ++t)
#pragma unroll
    for (int r = 0; r < 8; ++r) o_acc[t][r] = 0.0f;
#pragma unroll
  for (int r = 0; r < 8; ++r) { rm[r] = -1e30f; rs[r] = 0.0f; }

  const float scale = 0.125f;  // 1/sqrt(64)

  for (int kb0 = 0; kb0 < S; kb0 += 32) {
    if (kb0 + 32 < S) {  // prefetch next key/value block (global_prefetch_b8)
      __builtin_prefetch(K + ((size_t)(b * S + kb0 + 32 + l16) * kD) + h * kDH, 0, 0);
      __builtin_prefetch(V + ((size_t)(b * S + kb0 + 32 + half * 16 + l16) * kD) + h * kDH, 0, 0);
    }

    // ---- load ALL four K fragments first (one big load clause, one wait),
    //      then run the four score WMMAs back-to-back.
    v16bf bk[2][2];  // [sub][chunk]
#pragma unroll
    for (int sub = 0; sub < 2; ++sub) {
      const int key = kb0 + sub * 16 + l16;
      const __bf16* kptr = K + ((size_t)(b * S + key) * kD) + h * kDH;
#pragma unroll
      for (int c = 0; c < 2; ++c) {
        const v8bf lo = *(const v8bf*)(kptr + c * 32 + half * 16);
        const v8bf hi = *(const v8bf*)(kptr + c * 32 + half * 16 + 8);
#pragma unroll
        for (int e = 0; e < 8; ++e) { bk[sub][c][e] = lo[e]; bk[sub][c][8 + e] = hi[e]; }
      }
    }
    v8f sc[2];
#pragma unroll
    for (int sub = 0; sub < 2; ++sub) {
      v8f a0 = {};
      a0 = __builtin_amdgcn_wmma_f32_16x16x32_bf16(
          false, aq[0], false, bk[sub][0], (short)0, a0, false, false);
      a0 = __builtin_amdgcn_wmma_f32_16x16x32_bf16(
          false, aq[1], false, bk[sub][1], (short)0, a0, false, false);
      sc[sub] = a0;
    }

    // ---- online softmax over the 32 keys; DPP row reductions (no LDS)
#pragma unroll
    for (int r = 0; r < 8; ++r) {
      const float s0 = sc[0][r] * scale;
      const float s1 = sc[1][r] * scale;
      const float tm = row16_max(fmaxf(s0, s1));
      const float nm   = fmaxf(rm[r], tm);
      const float corr = __expf(rm[r] - nm);
      const float p0 = __expf(s0 - nm);
      const float p1 = __expf(s1 - nm);
      const float ps = row16_sum(p0 + p1);
      rs[r] = rs[r] * corr + ps;
      rm[r] = nm;
#pragma unroll
      for (int t = 0; t < 4; ++t) o_acc[t][r] *= corr;
      const int prow = r + half * 8;
      ldsP[wv][prow][l16]      = (__bf16)p0;
      ldsP[wv][prow][16 + l16] = (__bf16)p1;
    }
    __syncthreads();

    // ---- re-gather P as an A-fragment (C-layout -> A-layout via LDS)
    v16bf pa;
    {
      const v8bf lo = *(const v8bf*)(&ldsP[wv][l16][half * 16]);
      const v8bf hi = *(const v8bf*)(&ldsP[wv][l16][half * 16 + 8]);
#pragma unroll
      for (int e = 0; e < 8; ++e) { pa[e] = lo[e]; pa[8 + e] = hi[e]; }
    }
    __syncthreads();

    // ---- load ALL four V fragments, then four PV WMMAs
    v16bf bvv[4];
#pragma unroll
    for (int t = 0; t < 4; ++t) {
#pragma unroll
      for (int e = 0; e < 16; ++e) {
        const int key = kb0 + half * 16 + e;
        bvv[t][e] = V[((size_t)(b * S + key) * kD) + h * kDH + t * 16 + l16];
      }
    }
#pragma unroll
    for (int t = 0; t < 4; ++t) {
      o_acc[t] = __builtin_amdgcn_wmma_f32_16x16x32_bf16(
          false, pa, false, bvv[t], (short)0, o_acc[t], false, false);
    }
  }

  // ---- normalize and write heads (B,S,H,DH) flattened as (B,S,D)
#pragma unroll
  for (int t = 0; t < 4; ++t) {
#pragma unroll
    for (int r = 0; r < 8; ++r) {
      const int m   = qt * 16 + r + half * 8;
      const int col = h * kDH + t * 16 + l16;
      O[((size_t)(b * S + m) * kD) + col] = (__bf16)(o_acc[t][r] / rs[r]);
    }
  }
}

// --------------------------------------------------------- fused add+layernorm
__global__ void __launch_bounds__(256)
add_ln_kernel(const float* __restrict__ X, const float* __restrict__ R,
              const float* __restrict__ g, const float* __restrict__ be,
              float* __restrict__ outF, __bf16* __restrict__ outB) {
  __shared__ float sh[8];
  const int row = blockIdx.x;
  const int t = threadIdx.x;
  float v[4];
  float s = 0.f;
#pragma unroll
  for (int i = 0; i < 4; ++i) {
    const int d = t + i * 256;
    v[i] = X[(size_t)row * kD + d] + R[(size_t)row * kD + d];
    s += v[i];
  }
  s = row16_sum(s);
  s += __shfl_xor(s, 16, 32);
  if ((t & 31) == 0) sh[t >> 5] = s;
  __syncthreads();
  float tot = 0.f;
#pragma unroll
  for (int i = 0; i < 8; ++i) tot += sh[i];
  const float mean = tot * (1.0f / kD);
  __syncthreads();
  float s2 = 0.f;
#pragma unroll
  for (int i = 0; i < 4; ++i) { const float d0 = v[i] - mean; s2 += d0 * d0; }
  s2 = row16_sum(s2);
  s2 += __shfl_xor(s2, 16, 32);
  if ((t & 31) == 0) sh[t >> 5] = s2;
  __syncthreads();
  float tot2 = 0.f;
#pragma unroll
  for (int i = 0; i < 8; ++i) tot2 += sh[i];
  const float rstd = rsqrtf(tot2 * (1.0f / kD) + kEps);
#pragma unroll
  for (int i = 0; i < 4; ++i) {
    const int d = t + i * 256;
    const float y = (v[i] - mean) * rstd * g[d] + be[d];
    outF[(size_t)row * kD + d] = y;
    if (outB) outB[(size_t)row * kD + d] = (__bf16)y;
  }
}

// ------------------------------------------------------------------ launcher
extern "C" void kernel_launch(void* const* d_in, const int* in_sizes, int n_in,
                              void* d_out, int out_size, void* d_ws, size_t ws_size,
                              hipStream_t stream) {
  const int*   tokens = (const int*)d_in[0];
  const float* emb    = (const float*)d_in[1];
  const float* wq     = (const float*)d_in[2];
  const float* wk     = (const float*)d_in[3];
  const float* wvw    = (const float*)d_in[4];
  const float* wo     = (const float*)d_in[5];
  const float* fc1_w  = (const float*)d_in[6];
  const float* fc1_b  = (const float*)d_in[7];
  const float* fc2_w  = (const float*)d_in[8];
  const float* fc2_b  = (const float*)d_in[9];
  const float* ln1_s  = (const float*)d_in[10];
  const float* ln1_b  = (const float*)d_in[11];
  const float* ln2_s  = (const float*)d_in[12];
  const float* ln2_b  = (const float*)d_in[13];

  const int S  = in_sizes[0] / kB;   // 2048
  const int M  = kB * S;             // 4096 rows

  char* ws = (char*)d_ws;
  size_t off = 0;
  auto alloc = [&](size_t bytes) -> char* {
    char* p = ws + off;
    off += (bytes + 255) & ~(size_t)255;
    return p;
  };
  float*  x_f     = (float*) alloc((size_t)M * kD * 4);
  __bf16* x_b     = (__bf16*)alloc((size_t)M * kD * 2);
  __bf16* q_b     = (__bf16*)alloc((size_t)M * kD * 2);
  __bf16* k_b     = (__bf16*)alloc((size_t)M * kD * 2);
  __bf16* v_b     = (__bf16*)alloc((size_t)M * kD * 2);
  __bf16* heads_b = (__bf16*)alloc((size_t)M * kD * 2);
  float*  attn_f  = (float*) alloc((size_t)M * kD * 4);
  float*  h_f     = (float*) alloc((size_t)M * kD * 4);
  __bf16* h_b     = (__bf16*)alloc((size_t)M * kD * 2);
  __bf16* ff1_b   = (__bf16*)alloc((size_t)M * kFF * 2);
  float*  ff2_f   = (float*) alloc((size_t)M * kD * 4);
  __bf16* wq_b    = (__bf16*)alloc((size_t)kD * kD * 2);
  __bf16* wk_b    = (__bf16*)alloc((size_t)kD * kD * 2);
  __bf16* wv_b    = (__bf16*)alloc((size_t)kD * kD * 2);
  __bf16* wo_b    = (__bf16*)alloc((size_t)kD * kD * 2);
  __bf16* fc1w_b  = (__bf16*)alloc((size_t)kD * kFF * 2);
  __bf16* fc2w_b  = (__bf16*)alloc((size_t)kFF * kD * 2);

  // weight conversion
  f32_to_bf16_kernel<<<1024, 256, 0, stream>>>(wq,    wq_b,   kD * kD);
  f32_to_bf16_kernel<<<1024, 256, 0, stream>>>(wk,    wk_b,   kD * kD);
  f32_to_bf16_kernel<<<1024, 256, 0, stream>>>(wvw,   wv_b,   kD * kD);
  f32_to_bf16_kernel<<<1024, 256, 0, stream>>>(wo,    wo_b,   kD * kD);
  f32_to_bf16_kernel<<<1024, 256, 0, stream>>>(fc1_w, fc1w_b, kD * kFF);
  f32_to_bf16_kernel<<<1024, 256, 0, stream>>>(fc2_w, fc2w_b, kFF * kD);

  // embedding + positional encoding
  embed_pe_kernel<<<M, 256, 0, stream>>>(tokens, emb, x_f, x_b, S);

  // Q/K/V projections (bf16 out); block tile 64x128
  dim3 gQ(kD / 128, M / 64);
  gemm_wmma_bf16<__bf16, false, false><<<gQ, 256, 0, stream>>>(x_b, wq_b, q_b, nullptr, M, kD, kD);
  gemm_wmma_bf16<__bf16, false, false><<<gQ, 256, 0, stream>>>(x_b, wk_b, k_b, nullptr, M, kD, kD);
  gemm_wmma_bf16<__bf16, false, false><<<gQ, 256, 0, stream>>>(x_b, wv_b, v_b, nullptr, M, kD, kD);

  // fused attention -> heads
  attn_kernel<<<(kB * kH * (S / 16)) / 8, 256, 0, stream>>>(q_b, k_b, v_b, heads_b, S);

  // output projection (f32 out)
  gemm_wmma_bf16<float, false, false><<<gQ, 256, 0, stream>>>(heads_b, wo_b, attn_f, nullptr, M, kD, kD);

  // add & norm 1
  add_ln_kernel<<<M, 256, 0, stream>>>(x_f, attn_f, ln1_s, ln1_b, h_f, h_b);

  // FFN
  dim3 gF1(kFF / 128, M / 64);
  gemm_wmma_bf16<__bf16, true, true><<<gF1, 256, 0, stream>>>(h_b, fc1w_b, ff1_b, fc1_b, M, kFF, kD);
  dim3 gF2(kD / 128, M / 64);
  gemm_wmma_bf16<float, true, false><<<gF2, 256, 0, stream>>>(ff1_b, fc2w_b, ff2_f, fc2_b, M, kD, kFF);

  // add & norm 2 -> output
  add_ln_kernel<<<M, 256, 0, stream>>>(h_f, ff2_f, ln2_s, ln2_b, (float*)d_out, nullptr);
}
